// GQA_81131932221461
// MI455X (gfx1250) — matrix operand order repaired and checked by reading, not compiled, not run on hardware
//
#include <hip/hip_runtime.h>

// ---------------------------------------------------------------------------
// GQA attention block for MI455X (gfx1250, wave32, WMMA + TDM).
// Pipeline: rmsnorm(x)->bf16 | w->bf16 | WMMA GEMM qkv (double-buffered
// TDM-staged A tile in LDS, 4-wave blocks, 16x64 per wave) | rope+headnorm |
// flash-attn (double-buffered TDM-staged K/V in LDS, WMMA QK^T + online
// softmax + WMMA PV, P via LDS) | WMMA GEMM output proj -> f32 d_out.
// ---------------------------------------------------------------------------

#define EMBED_DIM    1024
#define NUM_HEADS    16
#define NUM_KV_HEADS 4
#define HEAD_DIM     64
#define SEQ_T        2048
#define BATCH        2
#define GQA_G        (NUM_HEADS / NUM_KV_HEADS)
#define KV_DIM       (NUM_KV_HEADS * HEAD_DIM)   // 256
#define EPS_F        1.1920929e-07f
#define KCHUNK       128                          // K staged per LDS tile

typedef __attribute__((ext_vector_type(16))) __bf16 v16bf;
typedef __attribute__((ext_vector_type(8)))  __bf16 v8bf;
typedef __attribute__((ext_vector_type(8)))  float  v8f;
typedef __attribute__((ext_vector_type(4)))  unsigned int v4u;
typedef __attribute__((ext_vector_type(8)))  int    v8i;
typedef __attribute__((ext_vector_type(4)))  int    v4i;

__device__ __forceinline__ unsigned short bfbits(float f) {
  union { float f; unsigned u; } in; in.f = f;
  unsigned r = in.u + 0x7FFFu + ((in.u >> 16) & 1u);   // round-to-nearest-even
  return (unsigned short)(r >> 16);
}
__device__ __forceinline__ __bf16 bits2bf(unsigned short s) {
  union { unsigned short s; __bf16 b; } u; u.s = s;
  return u.b;
}

#if __has_builtin(__builtin_amdgcn_tensor_load_to_lds)
#define HAVE_TDM 1
// 2D bf16 tile DMA: global[tile_d1 rows x tile_d0 cols, row stride stride0
// elements] -> LDS at byte offset ldsOff. D# per CDNA5 ISA 8.3/8.4.
__device__ __forceinline__ void tdm_load_tile_2d(unsigned ldsOff,
                                                 const unsigned short* gptr,
                                                 unsigned tensor_d0, unsigned tensor_d1,
                                                 unsigned tile_d0, unsigned tile_d1,
                                                 unsigned stride0) {
  unsigned long long ga = (unsigned long long)(size_t)gptr;
  v4u g0 = { 1u,                                   // count=1 (valid), no gather
             ldsOff,                               // lds_addr (bytes)
             (unsigned)(ga & 0xFFFFFFFFull),       // global_addr[31:0]
             (unsigned)((ga >> 32) & 0x01FFFFFFull) | (2u << 30) }; // [56:32]|type=2
  v8i g1 = { (int)(1u << 16),                                   // data_size=2B, mask=0
             (int)((tensor_d0 & 0xFFFFu) << 16),                // td0[15:0]
             (int)((tensor_d0 >> 16) | ((tensor_d1 & 0xFFFFu) << 16)),
             (int)((tensor_d1 >> 16) | (tile_d0 << 16)),
             (int)(tile_d1),                                    // tile_d1 | tile_d2=0
             (int)(stride0),                                    // stride0[31:0]
             0, 0 };                                            // stride0 hi, stride1
  v4i gz = { 0, 0, 0, 0 };
#if defined(__clang_major__) && __clang_major__ >= 23
  v8i gz8 = { 0, 0, 0, 0, 0, 0, 0, 0 };
  __builtin_amdgcn_tensor_load_to_lds(g0, g1, gz, gz, gz8, 0);
#else
  __builtin_amdgcn_tensor_load_to_lds(g0, g1, gz, gz, 0);
#endif
}
#endif

// ---------------- elementwise kernels --------------------------------------

__global__ void rmsnorm_x_kernel(const float* __restrict__ x,
                                 const float* __restrict__ w,
                                 unsigned short* __restrict__ xn) {
  const int row = blockIdx.x;            // B*T rows
  const int tid = threadIdx.x;           // 256 threads, 4 elems each
  const float* xr = x + (size_t)row * EMBED_DIM;
  float vals[4]; float ss = 0.0f;
#pragma unroll
  for (int i = 0; i < 4; ++i) { vals[i] = xr[tid + i * 256]; ss += vals[i] * vals[i]; }
  __shared__ float red[256];
  red[tid] = ss; __syncthreads();
  for (int o = 128; o > 0; o >>= 1) { if (tid < o) red[tid] += red[tid + o]; __syncthreads(); }
  const float r = rsqrtf(red[0] / (float)EMBED_DIM + EPS_F);
#pragma unroll
  for (int i = 0; i < 4; ++i) {
    const int c = tid + i * 256;
    xn[(size_t)row * EMBED_DIM + c] = bfbits(vals[i] * r * w[c]);
  }
}

__global__ void cast_f32_bf16(const float* __restrict__ in,
                              unsigned short* __restrict__ out, int n) {
  int i = blockIdx.x * blockDim.x + threadIdx.x;
  if (i < n) out[i] = bfbits(in[i]);
}

// rope + per-head rmsnorm; in: [B*T, nheads*64] f32 ; out: [B,nheads,T,64] bf16
__global__ void rope_norm_kernel(const float* __restrict__ inp,
                                 const float* __restrict__ w,
                                 unsigned short* __restrict__ out, int nheads) {
  const int t = blockIdx.x, h = blockIdx.y, b = blockIdx.z, d = threadIdx.x; // d in [0,64)
  const float* row = inp + ((size_t)(b * SEQ_T + t)) * (nheads * HEAD_DIM) + h * HEAD_DIM;
  const float xv = row[d];
  const int j = d & 31;
  const float invf = __powf(10000.0f, -(float)j / 32.0f);
  const float ang = (float)t * invf;
  const float c = __cosf(ang), s = __sinf(ang);
  const float other = (d < 32) ? -row[d + 32] : row[d - 32];
  const float y = xv * c + other * s;
  __shared__ float red[64];
  red[d] = y * y; __syncthreads();
  for (int o = 32; o > 0; o >>= 1) { if (d < o) red[d] += red[d + o]; __syncthreads(); }
  const float r = rsqrtf(red[0] / (float)HEAD_DIM + EPS_F);
  out[(((size_t)(b * nheads + h)) * SEQ_T + t) * HEAD_DIM + d] = bfbits(y * r * w[d]);
}

// v: [B*T, 4*64] f32 -> [B,4,T,64] bf16
__global__ void vcast_kernel(const float* __restrict__ vf,
                             unsigned short* __restrict__ vb) {
  const int t = blockIdx.x, h = blockIdx.y, b = blockIdx.z, d = threadIdx.x;
  vb[(((size_t)(b * NUM_KV_HEADS + h)) * SEQ_T + t) * HEAD_DIM + d] =
      bfbits(vf[((size_t)(b * SEQ_T + t)) * KV_DIM + h * HEAD_DIM + d]);
}

// ---------------- WMMA GEMM: C[M,N] = A[M,K](bf16) * W[N,K]^T(bf16) --------
// 128-thread blocks (4 waves); block tile = 16 (M) x 256 (N); each wave does
// 16x64 (4 accumulators) reusing one A fragment per 4 WMMAs. A is staged per
// 128-wide K-chunk into double-buffered LDS by the Tensor Data Mover (wave 0
// issues the DMA for chunk c+1 while all waves compute chunk c).

__global__ __launch_bounds__(128) void gemm_bf16_kernel(
    const unsigned short* __restrict__ A,
    const unsigned short* __restrict__ W,
    float* __restrict__ C, int M, int N, int K) {
  __shared__ __align__(16) unsigned short ldsA[2][16 * KCHUNK];
  const int tid = threadIdx.x;
  const int wv = tid >> 5;
  const int lane = tid & 31;
  const int half = lane >> 4;
  const int ln = lane & 15;
  const int m0 = blockIdx.y * 16;
  const int nw = blockIdx.x * 256 + wv * 64;

  const v8f vz = {};
  v8f acc[4];
#pragma unroll
  for (int t = 0; t < 4; ++t) acc[t] = vz;

  const int nc = K / KCHUNK;
#if defined(HAVE_TDM)
  if (wv == 0)
    tdm_load_tile_2d((unsigned)(size_t)&ldsA[0][0], A + (size_t)m0 * K,
                     (unsigned)K, (unsigned)M, KCHUNK, 16, (unsigned)K);
#endif
  for (int c = 0; c < nc; ++c) {
    const int k0 = c * KCHUNK;
    const unsigned short* atile = &ldsA[c & 1][0];
#if defined(HAVE_TDM)
    if (wv == 0) __builtin_amdgcn_s_wait_tensorcnt(0);  // chunk c resident
    __syncthreads();                                    // all waves past prior reads
    if (wv == 0 && c + 1 < nc)                          // async prefetch chunk c+1
      tdm_load_tile_2d((unsigned)(size_t)&ldsA[(c + 1) & 1][0],
                       A + (size_t)m0 * K + k0 + KCHUNK,
                       (unsigned)K, (unsigned)M, KCHUNK, 16, (unsigned)K);
#else
    __syncthreads();
    for (int cc = tid; cc < 16 * KCHUNK / 8; cc += 128) {   // 16B chunks
      const int row = cc >> 4;
      const int col = (cc & 15) * 8;
      *(uint4*)&ldsA[c & 1][row * KCHUNK + col] =
          *(const uint4*)&A[(size_t)(m0 + row) * K + k0 + col];
    }
    __syncthreads();
#endif

#pragma unroll
    for (int ks = 0; ks < KCHUNK / 32; ++ks) {
      // A fragment from LDS: lanes 0-15 hold K 0..15, 16-31 K 16..31
      union { v16bf v; v8bf h2[2]; } ua;
      ua.h2[0] = *(const v8bf*)&atile[ln * KCHUNK + ks * 32 + half * 16];
      ua.h2[1] = *(const v8bf*)&atile[ln * KCHUNK + ks * 32 + half * 16 + 8];
#pragma unroll
      for (int t = 0; t < 4; ++t) {
        union { v16bf v; v8bf h2[2]; } ub;   // W^T: lanes 0-15 K 0..15, 16-31 K 16..31
        const v8bf* pb = (const v8bf*)(W + (size_t)(nw + t * 16 + ln) * K +
                                       k0 + ks * 32 + half * 16);
        ub.h2[0] = pb[0]; ub.h2[1] = pb[1];
        acc[t] = __builtin_amdgcn_wmma_f32_16x16x32_bf16(false, ua.v, false, ub.v,
                                                         (short)0, acc[t], false, false);
      }
    }
  }
#pragma unroll
  for (int t = 0; t < 4; ++t)
#pragma unroll
    for (int r = 0; r < 8; ++r)
      C[(size_t)(m0 + r + half * 8) * N + nw + t * 16 + ln] = acc[t][r];
}

// ---------------- flash attention (one wave per 16-query tile) -------------
// q: [B,H,T,64] bf16 ; k,v: [B,Hkv,T,64] bf16 ; out: [B*T, H*64] bf16
// K/V 32-key blocks are TDM-staged into double-buffered LDS; the DMA for
// block j+1 overlaps the WMMA work on block j (single wave: s_wait_tensorcnt
// gates reads, s_wait_dscnt gates re-targeting the buffer read last round).

__global__ __launch_bounds__(32) void attn_fa_kernel(
    const unsigned short* __restrict__ qbf,
    const unsigned short* __restrict__ kbf,
    const unsigned short* __restrict__ vbf,
    unsigned short* __restrict__ obf) {
  const int qt = blockIdx.x, h = blockIdx.y, b = blockIdx.z;
  const int hkv = h / GQA_G;
  const int lane = threadIdx.x;
  const int half = lane >> 4;
  const int ln = lane & 15;
  const int q0 = qt * 16;

  __shared__ __align__(16) unsigned short plds[16][36];          // P transpose
  __shared__ __align__(16) unsigned short kls[2][32 * HEAD_DIM]; // K blocks
  __shared__ __align__(16) unsigned short vls[2][32 * HEAD_DIM]; // V blocks

  const unsigned short* qbase = qbf + (((size_t)(b * NUM_HEADS + h)) * SEQ_T + q0) * HEAD_DIM;
  const unsigned short* kb = kbf + ((size_t)(b * NUM_KV_HEADS + hkv)) * SEQ_T * HEAD_DIM;
  const unsigned short* vb = vbf + ((size_t)(b * NUM_KV_HEADS + hkv)) * SEQ_T * HEAD_DIM;

  // Q A-fragments: 16x64 as two k-steps of 32
  v16bf aq[2];
#pragma unroll
  for (int ks = 0; ks < 2; ++ks) {
    union { v16bf v; v8bf h2[2]; } u;
    u.h2[0] = *(const v8bf*)(qbase + (size_t)ln * HEAD_DIM + ks * 32 + half * 8);
    u.h2[1] = *(const v8bf*)(qbase + (size_t)ln * HEAD_DIM + ks * 32 + 16 + half * 8);
    aq[ks] = u.v;
  }

  const v8f vz = {};
  v8f accv[4];
#pragma unroll
  for (int a = 0; a < 4; ++a) accv[a] = vz;
  float mrow[8], lrow[8];
#pragma unroll
  for (int r = 0; r < 8; ++r) { mrow[r] = -3.0e38f; lrow[r] = 0.0f; }

  const int nkb = (q0 + 47) / 32;                 // 32-key blocks covering [0, q0+15]
#if defined(HAVE_TDM)
  tdm_load_tile_2d((unsigned)(size_t)&kls[0][0], kb, HEAD_DIM, SEQ_T, HEAD_DIM, 32, HEAD_DIM);
  tdm_load_tile_2d((unsigned)(size_t)&vls[0][0], vb, HEAD_DIM, SEQ_T, HEAD_DIM, 32, HEAD_DIM);
#endif
  for (int kb32 = 0; kb32 < nkb; ++kb32) {
    const int j0 = kb32 * 32;
    const int cur = kb32 & 1;
    const unsigned short* kcur = &kls[cur][0];
    const unsigned short* vcur = &vls[cur][0];
#if defined(HAVE_TDM)
    __builtin_amdgcn_s_wait_tensorcnt(0);          // block kb32 resident in LDS
    if (kb32 + 1 < nkb) {
      asm volatile("s_wait_dscnt 0x0" ::: "memory");  // prior reads of buf cur^1 done
      tdm_load_tile_2d((unsigned)(size_t)&kls[cur ^ 1][0],
                       kb + (size_t)(j0 + 32) * HEAD_DIM,
                       HEAD_DIM, SEQ_T, HEAD_DIM, 32, HEAD_DIM);
      tdm_load_tile_2d((unsigned)(size_t)&vls[cur ^ 1][0],
                       vb + (size_t)(j0 + 32) * HEAD_DIM,
                       HEAD_DIM, SEQ_T, HEAD_DIM, 32, HEAD_DIM);
    }
#else
    for (int cc = lane; cc < 32 * HEAD_DIM / 8; cc += 32) {    // 16B chunks
      *(uint4*)&kls[cur][cc * 8] = *(const uint4*)&kb[(size_t)j0 * HEAD_DIM + cc * 8];
      *(uint4*)&vls[cur][cc * 8] = *(const uint4*)&vb[(size_t)j0 * HEAD_DIM + cc * 8];
    }
    __syncthreads();
#endif

    // ---- S = Q K^T (two 16x16 tiles, each 2 chained WMMAs over hd=64) ----
    float sv[2][8];
#pragma unroll
    for (int jj = 0; jj < 2; ++jj) {
      v8f s = vz;
#pragma unroll
      for (int ks = 0; ks < 2; ++ks) {
        union { v16bf v; v8bf h2[2]; } ub;   // B = K^T slice: lane = key, elems = hd
        const v8bf* p = (const v8bf*)&kcur[(jj * 16 + ln) * HEAD_DIM + ks * 32 + half * 8];
        ub.h2[0] = p[0]; ub.h2[1] = p[2];
        s = __builtin_amdgcn_wmma_f32_16x16x32_bf16(false, aq[ks], false, ub.v,
                                                    (short)0, s, false, false);
      }
#pragma unroll
      for (int r = 0; r < 8; ++r) {
        const int m = r + half * 8;          // query row in tile
        const int n = j0 + jj * 16 + ln;     // key index
        float val = s[r] * 0.125f;           // 1/sqrt(64)
        if (n > q0 + m) val = -3.0e38f;      // causal mask
        sv[jj][r] = val;
      }
    }

    // ---- online softmax: rows live across the 16 lanes of each half ----
#pragma unroll
    for (int r = 0; r < 8; ++r) {
      float mx = fmaxf(sv[0][r], sv[1][r]);
#pragma unroll
      for (int o = 1; o < 16; o <<= 1) mx = fmaxf(mx, __shfl_xor(mx, o, 32));
      const float mnew = fmaxf(mrow[r], mx);
      const float corr = __expf(mrow[r] - mnew);
      const float p0 = __expf(sv[0][r] - mnew);
      const float p1 = __expf(sv[1][r] - mnew);
      sv[0][r] = p0; sv[1][r] = p1;
      float ls = p0 + p1;
#pragma unroll
      for (int o = 1; o < 16; o <<= 1) ls += __shfl_xor(ls, o, 32);
      lrow[r] = lrow[r] * corr + ls;
      mrow[r] = mnew;
#pragma unroll
      for (int a = 0; a < 4; ++a) accv[a][r] *= corr;
    }

    // ---- P: C-layout -> A-layout through LDS ----
    __syncthreads();
#pragma unroll
    for (int jj = 0; jj < 2; ++jj)
#pragma unroll
      for (int r = 0; r < 8; ++r)
        plds[r + half * 8][jj * 16 + ln] = bfbits(sv[jj][r]);
    __syncthreads();

    union { v16bf v; __bf16 el[16]; } up;
#pragma unroll
    for (int e = 0; e < 16; ++e) {
      const int k_local = ((e >> 3) << 4) + half * 8 + (e & 7);
      up.el[e] = bits2bf(plds[ln][k_local]);
    }
    __syncthreads();

    // ---- O += P (16x32) @ V (32x64): 4 WMMAs over hd column groups ----
#pragma unroll
    for (int a = 0; a < 4; ++a) {
      union { v16bf v; __bf16 el[16]; } uv;   // B = V block: lane = hd col, elems = key
#pragma unroll
      for (int e = 0; e < 16; ++e) {
        const int kk = ((e >> 3) << 4) + half * 8 + (e & 7);
        uv.el[e] = bits2bf(vcur[(size_t)kk * HEAD_DIM + a * 16 + ln]);
      }
      accv[a] = __builtin_amdgcn_wmma_f32_16x16x32_bf16(false, up.v, false, uv.v,
                                                        (short)0, accv[a], false, false);
    }
  }

  // ---- epilogue: O /= l, store bf16 into [B*T, H*64] ----
#pragma unroll
  for (int r = 0; r < 8; ++r) {
    const float inv = 1.0f / lrow[r];
    const int m = r + half * 8;
    unsigned short* orow = obf + ((size_t)(b * SEQ_T) + q0 + m) * EMBED_DIM + h * HEAD_DIM;
#pragma unroll
    for (int a = 0; a < 4; ++a)
      orow[a * 16 + ln] = bfbits(accv[a][r] * inv);
  }
}

// ---------------- host launch ----------------------------------------------

extern "C" void kernel_launch(void* const* d_in, const int* in_sizes, int n_in,
                              void* d_out, int out_size, void* d_ws, size_t ws_size,
                              hipStream_t stream) {
  (void)in_sizes; (void)n_in; (void)out_size; (void)ws_size;
  const float* x      = (const float*)d_in[0];
  const float* wq     = (const float*)d_in[1];
  const float* wk     = (const float*)d_in[2];
  const float* wv     = (const float*)d_in[3];
  const float* wo     = (const float*)d_in[4];
  const float* norm_w = (const float*)d_in[5];
  const float* qn_w   = (const float*)d_in[6];
  const float* kn_w   = (const float*)d_in[7];
  float* out = (float*)d_out;

  char* ws = (char*)d_ws;
  size_t off = 0;
  auto alloc = [&](size_t bytes) {
    void* p = ws + off;
    off += (bytes + 255) & ~(size_t)255;
    return p;
  };

  const size_t M = (size_t)BATCH * SEQ_T;  // 4096
  unsigned short* xn    = (unsigned short*)alloc(M * EMBED_DIM * 2);
  unsigned short* wqb   = (unsigned short*)alloc((size_t)EMBED_DIM * EMBED_DIM * 2);
  unsigned short* wkb   = (unsigned short*)alloc((size_t)KV_DIM * EMBED_DIM * 2);
  unsigned short* wvb   = (unsigned short*)alloc((size_t)KV_DIM * EMBED_DIM * 2);
  unsigned short* wob   = (unsigned short*)alloc((size_t)EMBED_DIM * EMBED_DIM * 2);
  float*          qf    = (float*)alloc(M * EMBED_DIM * 4);
  float*          kf    = (float*)alloc(M * KV_DIM * 4);
  float*          vf    = (float*)alloc(M * KV_DIM * 4);
  unsigned short* qb    = (unsigned short*)alloc(M * EMBED_DIM * 2);
  unsigned short* kbuf  = (unsigned short*)alloc(M * KV_DIM * 2);
  unsigned short* vbuf  = (unsigned short*)alloc(M * KV_DIM * 2);
  unsigned short* attnb = (unsigned short*)alloc(M * EMBED_DIM * 2);

  // 1) RMSNorm(x) -> bf16
  rmsnorm_x_kernel<<<dim3((unsigned)M), dim3(256), 0, stream>>>(x, norm_w, xn);

  // 2) weights -> bf16
  const int n1 = EMBED_DIM * EMBED_DIM, n2 = KV_DIM * EMBED_DIM;
  cast_f32_bf16<<<dim3((n1 + 255) / 256), dim3(256), 0, stream>>>(wq, wqb, n1);
  cast_f32_bf16<<<dim3((n2 + 255) / 256), dim3(256), 0, stream>>>(wk, wkb, n2);
  cast_f32_bf16<<<dim3((n2 + 255) / 256), dim3(256), 0, stream>>>(wv, wvb, n2);
  cast_f32_bf16<<<dim3((n1 + 255) / 256), dim3(256), 0, stream>>>(wo, wob, n1);

  // 3) projections (WMMA, double-buffered TDM-staged A tiles)
  gemm_bf16_kernel<<<dim3(EMBED_DIM / 256, (unsigned)(M / 16)), dim3(128), 0, stream>>>(
      xn, wqb, qf, (int)M, EMBED_DIM, EMBED_DIM);
  gemm_bf16_kernel<<<dim3(KV_DIM / 256, (unsigned)(M / 16)), dim3(128), 0, stream>>>(
      xn, wkb, kf, (int)M, KV_DIM, EMBED_DIM);
  gemm_bf16_kernel<<<dim3(KV_DIM / 256, (unsigned)(M / 16)), dim3(128), 0, stream>>>(
      xn, wvb, vf, (int)M, KV_DIM, EMBED_DIM);

  // 4) RoPE + per-head RMSNorm -> bf16 [B,h,T,64]; V transpose-cast
  rope_norm_kernel<<<dim3(SEQ_T, NUM_HEADS, BATCH), dim3(64), 0, stream>>>(qf, qn_w, qb, NUM_HEADS);
  rope_norm_kernel<<<dim3(SEQ_T, NUM_KV_HEADS, BATCH), dim3(64), 0, stream>>>(kf, kn_w, kbuf, NUM_KV_HEADS);
  vcast_kernel<<<dim3(SEQ_T, NUM_KV_HEADS, BATCH), dim3(64), 0, stream>>>(vf, vbuf);

  // 5) causal GQA flash attention (WMMA, double-buffered TDM K/V)
  attn_fa_kernel<<<dim3(SEQ_T / 16, NUM_HEADS, BATCH), dim3(32), 0, stream>>>(qb, kbuf, vbuf, attnb);

  // 6) output projection (WMMA) -> f32 d_out
  gemm_bf16_kernel<<<dim3(EMBED_DIM / 256, (unsigned)(M / 16)), dim3(128), 0, stream>>>(
      attnb, wob, out, (int)M, EMBED_DIM, EMBED_DIM);
}
